// HODLRLoss_49082886259365
// MI455X (gfx1250) — matrix operand order repaired and checked by reading, not compiled, not run on hardware
//
#include <hip/hip_runtime.h>
#include <hip/hip_bf16.h>
#include <math.h>

// ---------------------------------------------------------------------------
// HODLR apply + sparse matvec + MSE loss for MI455X (gfx1250, wave32).
// Memory-bound workload (~570 MB/call, AI ~0.5 flop/B -> ~25us floor @23.3TB/s)
// so everything stays f32 and single-pass. Leaf 32x32 blocks use
// V_WMMA_F32_16X16X4_F32 (f32 matrix cores) with L staged into LDS via
// GLOBAL_LOAD_ASYNC_TO_LDS_B128 (ASYNCcnt path); the streaming low-rank
// levels use b128 loads + global_prefetch_b8.
// ---------------------------------------------------------------------------

typedef float    v2f  __attribute__((ext_vector_type(2)));
typedef float    v8f  __attribute__((ext_vector_type(8)));
typedef _Float16 v16h __attribute__((ext_vector_type(16)));

// async-to-LDS builtin operand types (from hipcc diagnostic: param0 is
// "int __vector(4) __device__ *", i.e. v4i in address_space(1))
typedef int v4i_t __attribute__((vector_size(16)));
typedef __attribute__((address_space(1))) v4i_t* as1_v4i_ptr;
typedef __attribute__((address_space(3))) v4i_t* as3_v4i_ptr;

#define LEAF 32
#define WAVES_PER_BLOCK 8
// per-wave LDS region: L tile 32x32 (stride 33 to dodge bank conflicts) + x,t,y
#define LDS_WAVE_STRIDE (32 * 33 + 96)

#if __has_builtin(__builtin_amdgcn_global_load_async_to_lds_b128)
#define HAVE_ASYNC_LDS 1
#else
#define HAVE_ASYNC_LDS 0
#endif

__device__ __forceinline__ unsigned int pcg_hash(unsigned int x) {
  x = x * 747796405u + 2891336453u;
  unsigned int w = ((x >> ((x >> 28u) + 4u)) ^ x) * 277803737u;
  return (w >> 22u) ^ w;
}

// ---------------------------------------------------------------------------
__global__ void k_zero(float* __restrict__ p, int n) {
  int i = blockIdx.x * blockDim.x + threadIdx.x;
  if (i < n) p[i] = 0.0f;
}

// z ~ N(0,1) stand-in (hash + Box-Muller); JAX threefry not reproducible here.
__global__ void k_genz(float* __restrict__ z, int N) {
  int i = blockIdx.x * blockDim.x + threadIdx.x;
  if (i < N) {
    unsigned int h1 = pcg_hash(2u * (unsigned)i + 1u);
    unsigned int h2 = pcg_hash(2u * (unsigned)i + 2u);
    float u1 = ((float)h1 + 1.0f) * (1.0f / 4294967296.0f);
    float u2 = ((float)h2) * (1.0f / 4294967296.0f);
    z[i] = sqrtf(-2.0f * logf(u1)) * __cosf(6.28318530718f * u2);
  }
}

// ---------------------------------------------------------------------------
// Leaf stage: per 32x32 block k:  w = L * (L^T * x) + 1e-4 * x
// One wave per block, 8 blocks per 256-thread workgroup.
// Each 32-half of the result = 8x accumulating WMMA f32 16x16x4 with x
// replicated across all 16 N-columns; column 0 of D carries the GEMV result.
// ---------------------------------------------------------------------------
__global__ __launch_bounds__(256) void k_leaf(const float* __restrict__ L,
                                              const float* __restrict__ z,
                                              float* __restrict__ w,
                                              int nblk) {
  __shared__ float lds[WAVES_PER_BLOCK * LDS_WAVE_STRIDE];
  const int tid  = threadIdx.x;
  const int wave = tid >> 5;
  const int lane = tid & 31;
  int blk = blockIdx.x * WAVES_PER_BLOCK + wave;
  const bool live = (blk < nblk);
  if (!live) blk = nblk - 1;  // clamp so EXEC stays all-ones at every WMMA

  float* Ls = lds + wave * LDS_WAVE_STRIDE;  // [32][33]
  float* xs = Ls + 32 * 33;                  // [32]
  float* ts = xs + 32;                       // [32] t = L^T x
  float* ys = ts + 32;                       // [32] y = L t

  const float* Lg = L + (size_t)blk * (LEAF * LEAF);

#if HAVE_ASYNC_LDS
  // CDNA5 async path: memory -> LDS directly (no VGPR round trip), per-lane
  // LDS destinations preserve the stride-33 padded layout. ASYNCcnt-tracked.
  for (int i = 0; i < 8; ++i) {
    int idx = lane + 32 * i;                 // 16B chunk index 0..255
    int row = idx >> 3, c4 = (idx & 7) << 2;
    __builtin_amdgcn_global_load_async_to_lds_b128(
        (as1_v4i_ptr)(Lg + (size_t)idx * 4),
        (as3_v4i_ptr)&Ls[row * 33 + c4], 0, 0);
  }
  xs[lane] = z[(size_t)blk * LEAF + lane];
#if __has_builtin(__builtin_amdgcn_s_wait_asynccnt)
  __builtin_amdgcn_s_wait_asynccnt(0);
#else
  asm volatile("s_wait_asynccnt 0" ::: "memory");
#endif
#else
  // fallback: classic b128 load + ds_store staging, fully coalesced
  for (int i = 0; i < 8; ++i) {
    int idx = lane + 32 * i;                 // float4 index 0..255
    float4 v = ((const float4*)Lg)[idx];
    int row = idx >> 3, c4 = (idx & 7) << 2;
    float* d = &Ls[row * 33 + c4];
    d[0] = v.x; d[1] = v.y; d[2] = v.z; d[3] = v.w;
  }
  xs[lane] = z[(size_t)blk * LEAF + lane];
#endif
  __syncthreads();

  // A-fragment lane mapping (f32 16x4): lanes 0-15 -> M=lane, K={k0,k0+1};
  // lanes 16-31 -> M=lane-16, K={k0+2,k0+3}
  const int m    = lane & 15;
  const int koff = (lane >> 4) << 1;

  // ---- stage 1: t = L^T x ----
  for (int h = 0; h < 2; ++h) {
    v8f acc = {0.f, 0.f, 0.f, 0.f, 0.f, 0.f, 0.f, 0.f};
#if __has_builtin(__builtin_amdgcn_wmma_f32_16x16x4_f32)
    for (int j = 0; j < 8; ++j) {
      const int k0 = j * 4 + koff;
      v2f a, b;
      a.x = Ls[(k0 + 0) * 33 + h * 16 + m];   // A[m][k] = L^T[m][k] = L[k][m]
      a.y = Ls[(k0 + 1) * 33 + h * 16 + m];
      b.x = xs[k0 + 0];                       // B[k][n] = x[k]  (replicated in n)
      b.y = xs[k0 + 1];
      acc = __builtin_amdgcn_wmma_f32_16x16x4_f32(false, a, false, b,
                                                  (short)0, acc, false, false);
    }
#else
    {  // fallback: single K=32 f16 WMMA (codegen-confirmed builtin)
      v16h a, b;
      for (int e = 0; e < 16; ++e) {
        int ka = e + ((e >= 8) ? 8 : 0) + ((lane >= 16) ? 8 : 0);
        int kb = e + ((e >= 8) ? 8 : 0) + ((lane >= 16) ? 8 : 0);
        a[e] = (_Float16)Ls[ka * 33 + h * 16 + m];
        b[e] = (_Float16)xs[kb];
      }
      acc = __builtin_amdgcn_wmma_f32_16x16x32_f16(false, a, false, b,
                                                   (short)0, acc, false, false);
    }
#endif
    if (m == 0) {  // column n=0: lane 0 holds rows 0..7, lane 16 rows 8..15
      const int base = h * 16 + (lane >> 4) * 8;
      for (int v = 0; v < 8; ++v) ts[base + v] = acc[v];
    }
  }
  __syncthreads();

  // ---- stage 2: y = L t ----
  for (int h = 0; h < 2; ++h) {
    v8f acc = {0.f, 0.f, 0.f, 0.f, 0.f, 0.f, 0.f, 0.f};
#if __has_builtin(__builtin_amdgcn_wmma_f32_16x16x4_f32)
    for (int j = 0; j < 8; ++j) {
      const int k0 = j * 4 + koff;
      v2f a, b;
      a.x = Ls[(h * 16 + m) * 33 + k0 + 0];   // A[m][k] = L[m][k]
      a.y = Ls[(h * 16 + m) * 33 + k0 + 1];
      b.x = ts[k0 + 0];
      b.y = ts[k0 + 1];
      acc = __builtin_amdgcn_wmma_f32_16x16x4_f32(false, a, false, b,
                                                  (short)0, acc, false, false);
    }
#else
    {
      v16h a, b;
      for (int e = 0; e < 16; ++e) {
        int ka = e + ((e >= 8) ? 8 : 0) + ((lane >= 16) ? 8 : 0);
        int kb = e + ((e >= 8) ? 8 : 0) + ((lane >= 16) ? 8 : 0);
        a[e] = (_Float16)Ls[(h * 16 + m) * 33 + ka];
        b[e] = (_Float16)ts[kb];
      }
      acc = __builtin_amdgcn_wmma_f32_16x16x32_f16(false, a, false, b,
                                                   (short)0, acc, false, false);
    }
#endif
    if (m == 0) {
      const int base = h * 16 + (lane >> 4) * 8;
      for (int v = 0; v < 8; ++v) ys[base + v] = acc[v];
    }
  }
  __syncthreads();

  if (live) w[(size_t)blk * LEAF + lane] = ys[lane] + 1e-4f * xs[lane];
}

// ---------------------------------------------------------------------------
// Level phase A: s[li,pair,side] = v_side^T @ x_side  (rank 64 dot products).
// One workgroup per 512 rows (all block sizes are multiples of 512).
// float4 lanes: 16 lanes x 16B = one full 256B row per row-group -> b128
// loads, 4x fewer VMEM instructions for the same 268 MB stream.
// ---------------------------------------------------------------------------
__global__ __launch_bounds__(256) void k_lvlA(const float* __restrict__ fv,
                                              const float* __restrict__ z,
                                              float* __restrict__ s,
                                              int N) {
  __shared__ float4 red[256];
  const int bpl = N >> 9;  // 512-row chunks per level
  const int li  = blockIdx.x / bpl;
  const int g0  = (blockIdx.x % bpl) << 9;
  const int Bl  = N >> (li + 1);
  const int p   = g0 / (Bl << 1);
  const int sd  = (g0 / Bl) & 1;
  const int sbase = 64 * ((2 << li) - 2) + (p * 2 + sd) * 64;

  const float* vb = fv + (size_t)li * (size_t)N * 64;
  const int q  = threadIdx.x & 15;   // column quad: cols 4q..4q+3
  const int rg = threadIdx.x >> 4;   // row group 0..15
  float4 a4 = {0.f, 0.f, 0.f, 0.f};
  for (int r = g0 + rg; r < g0 + 512; r += 16) {
    const float4* row4 = (const float4*)(vb + (size_t)r * 64);
    __builtin_prefetch(vb + (size_t)(r + 64) * 64 + 4 * q, 0, 1);  // global_prefetch_b8
    float4 v = row4[q];
    float zr = z[r];
    a4.x += v.x * zr; a4.y += v.y * zr; a4.z += v.z * zr; a4.w += v.w * zr;
  }
  red[threadIdx.x] = a4;
  __syncthreads();
  if (threadIdx.x < 16) {
    float4 t = {0.f, 0.f, 0.f, 0.f};
    for (int g = 0; g < 16; ++g) {
      float4 v = red[g * 16 + threadIdx.x];
      t.x += v.x; t.y += v.y; t.z += v.z; t.w += v.w;
    }
    atomicAdd(&s[sbase + 4 * threadIdx.x + 0], t.x);
    atomicAdd(&s[sbase + 4 * threadIdx.x + 1], t.y);
    atomicAdd(&s[sbase + 4 * threadIdx.x + 2], t.z);
    atomicAdd(&s[sbase + 4 * threadIdx.x + 3], t.w);
  }
}

// ---------------------------------------------------------------------------
// Level phase B: w[rows of side] += u_side @ s[li,pair,other-side]
// ---------------------------------------------------------------------------
__global__ __launch_bounds__(256) void k_lvlB(const float* __restrict__ fu,
                                              const float* __restrict__ s,
                                              float* __restrict__ w,
                                              int N) {
  __shared__ float so[64];
  const int bpl = N >> 9;
  const int li  = blockIdx.x / bpl;
  const int g0  = (blockIdx.x % bpl) << 9;
  const int Bl  = N >> (li + 1);
  const int p   = g0 / (Bl << 1);
  const int sd  = (g0 / Bl) & 1;
  const int sbase = 64 * ((2 << li) - 2) + (p * 2 + (1 - sd)) * 64;  // other side
  if (threadIdx.x < 64) so[threadIdx.x] = s[sbase + threadIdx.x];
  __syncthreads();

  const float* ub = fu + (size_t)li * (size_t)N * 64;
  for (int rr = 0; rr < 2; ++rr) {
    const int g = g0 + rr * 256 + threadIdx.x;
    const float4* u4 = (const float4*)(ub + (size_t)g * 64);
    __builtin_prefetch(ub + (size_t)(g + 512) * 64, 0, 1);
    float d = 0.0f;
    for (int i = 0; i < 16; ++i) {
      float4 u = u4[i];
      d += u.x * so[4 * i] + u.y * so[4 * i + 1] +
           u.z * so[4 * i + 2] + u.w * so[4 * i + 3];
    }
    atomicAdd(&w[g], d);
  }
}

// ---------------------------------------------------------------------------
// Sparse:  y[row] += A_val * w[col]   (segment_sum via f32 atomics)
// ---------------------------------------------------------------------------
__global__ void k_sparse(const long long* __restrict__ idx,
                         const float* __restrict__ vals,
                         const float* __restrict__ w,
                         float* __restrict__ yo, int nnz) {
  int i = blockIdx.x * blockDim.x + threadIdx.x;
  if (i < nnz) {
    long long r = idx[i];
    long long c = idx[(size_t)nnz + i];
    atomicAdd(&yo[r], vals[i] * w[c]);
  }
}

// ---------------------------------------------------------------------------
// Loss = mean((y + tail(w) - z)^2); tail add fused in to save a pass.
// ---------------------------------------------------------------------------
__global__ __launch_bounds__(256) void k_loss(const float* __restrict__ yo,
                                              const float* __restrict__ w,
                                              const float* __restrict__ z,
                                              const int* __restrict__ nreal_p,
                                              float* __restrict__ acc, int N) {
  __shared__ float red[256];
  const int nreal = *nreal_p;
  float sum = 0.0f;
  for (int i = blockIdx.x * blockDim.x + threadIdx.x; i < N;
       i += gridDim.x * blockDim.x) {
    float y = yo[i];
    if (i >= nreal) y += w[i];
    float d = y - z[i];
    sum += d * d;
  }
  red[threadIdx.x] = sum;
  __syncthreads();
  for (int off = 128; off > 0; off >>= 1) {
    if (threadIdx.x < off) red[threadIdx.x] += red[threadIdx.x + off];
    __syncthreads();
  }
  if (threadIdx.x == 0) atomicAdd(acc, red[0]);
}

__global__ void k_final(const float* __restrict__ acc,
                        float* __restrict__ out, float invN) {
  out[0] = acc[0] * invN;
}

// ---------------------------------------------------------------------------
extern "C" void kernel_launch(void* const* d_in, const int* in_sizes, int n_in,
                              void* d_out, int out_size, void* d_ws,
                              size_t ws_size, hipStream_t stream) {
  const float*     leaf = (const float*)d_in[0];
  const float*     fu   = (const float*)d_in[1];
  const float*     fv   = (const float*)d_in[2];
  const long long* Aidx = (const long long*)d_in[3];  // int64 in reference
  const float*     Aval = (const float*)d_in[4];
  const int*       nreal = (const int*)d_in[6];       // device scalar

  const int N      = in_sizes[0] / 32;        // leaf_blocks holds N*32 floats
  const int nblk   = N / 32;
  const int levels = in_sizes[1] / (N * 64);  // factors_u = levels*N*64
  const int nnz    = in_sizes[4];
  const int S      = 64 * ((2 << levels) - 2);  // total s-vector floats

  // workspace layout (floats): z | w | yo | s | acc
  float* z   = (float*)d_ws;
  float* w   = z + N;
  float* yo  = w + N;
  float* s   = yo + N;
  float* acc = s + S;

  const int zcount = N + S + 1;  // zero yo, s, acc (contiguous)
  k_zero<<<(zcount + 255) / 256, 256, 0, stream>>>(yo, zcount);
  k_genz<<<(N + 255) / 256, 256, 0, stream>>>(z, N);
  k_leaf<<<(nblk + WAVES_PER_BLOCK - 1) / WAVES_PER_BLOCK, 256, 0, stream>>>(
      leaf, z, w, nblk);
  const int bpl = N >> 9;
  k_lvlA<<<levels * bpl, 256, 0, stream>>>(fv, z, s, N);
  k_lvlB<<<levels * bpl, 256, 0, stream>>>(fu, s, w, N);
  k_sparse<<<(nnz + 255) / 256, 256, 0, stream>>>(Aidx, Aval, w, yo, nnz);
  k_loss<<<1024, 256, 0, stream>>>(yo, w, z, nreal, acc, N);
  k_final<<<1, 1, 0, stream>>>(acc, (float*)d_out, 1.0f / (float)N);
}
